// GraphAttentionLayer_13073880449100
// MI455X (gfx1250) — matrix operand order repaired and checked by reading, not compiled, not run on hardware
//
#include <hip/hip_runtime.h>

#define N_NODES 8192
#define IN_F    512
#define OUT_F   256
#define ALPHA   0.2f
#define NEG_BIG -9.0e15f

typedef __attribute__((ext_vector_type(2)))  float    v2f;
typedef __attribute__((ext_vector_type(8)))  float    v8f;
typedef __attribute__((ext_vector_type(8)))  _Float16 v8h;
typedef __attribute__((ext_vector_type(16))) _Float16 v16h;

// ---------------------------------------------------------------------------
// Kernel 1: h = x @ W  via f32 WMMA 16x16x4.  Also emits hT (f16, transposed)
// for the later f16 WMMA pass.  Grid: 512 blocks x 512 threads (16 waves).
// Each wave owns one 16x16 C tile: block covers 16 rows x 256 cols.
// ---------------------------------------------------------------------------
__global__ __launch_bounds__(512) void k_gemm_h(
    const float* __restrict__ x, const float* __restrict__ W,
    float* __restrict__ h, _Float16* __restrict__ hT)
{
    const int lane = threadIdx.x & 31;
    const int wv   = threadIdx.x >> 5;
    const int row_base = blockIdx.x * 16;
    const int col_base = wv * 16;
    const int r = lane & 15;
    const int g = lane >> 4;          // lane half

    const float* xrow = x + (size_t)(row_base + r) * IN_F;
    const int n = col_base + r;       // B column for this lane

    v8f acc = {};
    #pragma unroll 4
    for (int k0 = 0; k0 < IN_F; k0 += 4) {
        // A (16x4 f32): V0 = K=2g, V1 = K=2g+1 for this lane half
        v2f a = *(const v2f*)(xrow + k0 + 2 * g);
        // B (4x16 f32): V0 = K=2g, V1 = K=2g+1, lane -> column n
        v2f b;
        b.x = W[(size_t)(k0 + 2 * g)     * OUT_F + n];
        b.y = W[(size_t)(k0 + 2 * g + 1) * OUT_F + n];
        acc = __builtin_amdgcn_wmma_f32_16x16x4_f32(
            false, a, false, b, (short)0, acc, false, false);
    }

    // C tile: VGPR v holds row (v + 8g), col = col_base + r
    #pragma unroll
    for (int v = 0; v < 8; ++v) {
        const int rr = row_base + v + (g << 3);
        const float val = acc[v];
        h [(size_t)rr * OUT_F + col_base + r] = val;
        hT[(size_t)(col_base + r) * N_NODES + rr] = (_Float16)val;
    }
}

// ---------------------------------------------------------------------------
// Kernel 2: attn_self = h @ a_self, attn_neighs = h @ a_neighs.
// One wave per row; lane covers 8 contiguous features.
// ---------------------------------------------------------------------------
__global__ __launch_bounds__(256) void k_attn(
    const float* __restrict__ h, const float* __restrict__ a_self,
    const float* __restrict__ a_neighs,
    float* __restrict__ out_s, float* __restrict__ out_n)
{
    const int lane = threadIdx.x & 31;
    const int wv   = threadIdx.x >> 5;
    const int row  = blockIdx.x * 8 + wv;

    const float4* hp  = (const float4*)(h + (size_t)row * OUT_F + lane * 8);
    const float4* asp = (const float4*)(a_self   + lane * 8);
    const float4* anp = (const float4*)(a_neighs + lane * 8);
    float4 h0 = hp[0],  h1 = hp[1];
    float4 s0 = asp[0], s1 = asp[1];
    float4 n0 = anp[0], n1 = anp[1];

    float ds = h0.x*s0.x + h0.y*s0.y + h0.z*s0.z + h0.w*s0.w
             + h1.x*s1.x + h1.y*s1.y + h1.z*s1.z + h1.w*s1.w;
    float dn = h0.x*n0.x + h0.y*n0.y + h0.z*n0.z + h0.w*n0.w
             + h1.x*n1.x + h1.y*n1.y + h1.z*n1.z + h1.w*n1.w;

    #pragma unroll
    for (int off = 16; off > 0; off >>= 1) {
        ds += __shfl_xor(ds, off, 32);
        dn += __shfl_xor(dn, off, 32);
    }
    if (lane == 0) { out_s[row] = ds; out_n[row] = dn; }
}

// ---------------------------------------------------------------------------
// Kernel 3: per-row online softmax stats (max, 1/sum) over masked leaky scores.
// One block (256 threads) per row; streams M and adj once.
// ---------------------------------------------------------------------------
__global__ __launch_bounds__(256) void k_stats(
    const float* __restrict__ M, const float* __restrict__ adj,
    const float* __restrict__ a_s, const float* __restrict__ a_n,
    float* __restrict__ rmax, float* __restrict__ rinv)
{
    const int row = blockIdx.x;
    const int t   = threadIdx.x;
    const float as = a_s[row];
    const float* Mr = M   + (size_t)row * N_NODES;
    const float* Ar = adj + (size_t)row * N_NODES;

    float m = -3.0e38f, s = 0.0f;
    auto upd = [&](float Mv, float Av, float nv) {
        float sc = (as + nv) * Mv;
        sc = sc >= 0.0f ? sc : ALPHA * sc;
        sc = Av > 0.0f ? sc : NEG_BIG;
        float nm = fmaxf(m, sc);
        s = s * __expf(m - nm) + __expf(sc - nm);
        m = nm;
    };

    #pragma unroll 2
    for (int i = 0; i < 8; ++i) {
        const int k = i * 1024 + t * 4;
        float4 m4 = *(const float4*)(Mr + k);
        float4 a4 = *(const float4*)(Ar + k);
        float4 n4 = *(const float4*)(a_n + k);
        upd(m4.x, a4.x, n4.x);
        upd(m4.y, a4.y, n4.y);
        upd(m4.z, a4.z, n4.z);
        upd(m4.w, a4.w, n4.w);
    }

    __shared__ float sm[256], ss[256];
    sm[t] = m; ss[t] = s;
    __syncthreads();
    for (int off = 128; off > 0; off >>= 1) {
        if (t < off) {
            float m1 = sm[t], s1 = ss[t];
            float m2 = sm[t + off], s2 = ss[t + off];
            float nm = fmaxf(m1, m2);
            ss[t] = s1 * __expf(m1 - nm) + s2 * __expf(m2 - nm);
            sm[t] = nm;
        }
        __syncthreads();
    }
    if (t == 0) { rmax[row] = sm[0]; rinv[row] = 1.0f / ss[0]; }
}

// ---------------------------------------------------------------------------
// Kernel 4: fused  out = elu( softmax(scores) @ h )  via f16 WMMA 16x16x32.
// Block = 16 rows x 256 cols (16 waves, one 16x16 C tile each).
// Per 512-wide K chunk: wave w recomputes probability row w into LDS
// (f16, padded rows); after barrier all waves run 16 WMMA K-steps,
// loading A from LDS and B from L2-resident transposed h.
// ---------------------------------------------------------------------------
__global__ __launch_bounds__(512) void k_fused(
    const float* __restrict__ M, const float* __restrict__ adj,
    const _Float16* __restrict__ hT,
    const float* __restrict__ a_s, const float* __restrict__ a_n,
    const float* __restrict__ rmax, const float* __restrict__ rinv,
    float* __restrict__ out)
{
    __shared__ _Float16 P[2][16][520];   // 512 data + 8 pad halves per row

    const int lane = threadIdx.x & 31;
    const int wv   = threadIdx.x >> 5;
    const int row_base = blockIdx.x * 16;
    const int r = lane & 15;
    const int g = lane >> 4;
    const int col = wv * 16 + r;

    // producer state: this wave produces P row `wv`
    const int prow   = row_base + wv;
    const float as   = a_s[prow];
    const float pm   = rmax[prow];
    const float pinv = rinv[prow];
    const float* Mr = M   + (size_t)prow * N_NODES;
    const float* Ar = adj + (size_t)prow * N_NODES;

    const _Float16* hTc = hT + (size_t)col * N_NODES;

    auto prob = [&](float Mv, float Av, float nv) -> float {
        float sc = (as + nv) * Mv;
        sc = sc >= 0.0f ? sc : ALPHA * sc;
        sc = Av > 0.0f ? sc : NEG_BIG;
        return __expf(sc - pm) * pinv;   // masked -> exp(<<0) -> 0
    };

    auto produce = [&](int c) {
        const int buf = c & 1;
        const int k = (c << 9) + lane * 16;   // lane covers 16 contiguous k
        _Float16* dst = &P[buf][wv][lane * 16];
        if (c < 15) {   // warm next chunk of the M/adj stream
            __builtin_prefetch(Mr + k + 512, 0, 0);
            __builtin_prefetch(Ar + k + 512, 0, 0);
        }
        v8h pv0, pv1;
        #pragma unroll
        for (int j = 0; j < 4; ++j) {
            float4 m4 = *(const float4*)(Mr + k + j * 4);
            float4 a4 = *(const float4*)(Ar + k + j * 4);
            float4 n4 = *(const float4*)(a_n + k + j * 4);
            float p0 = prob(m4.x, a4.x, n4.x);
            float p1 = prob(m4.y, a4.y, n4.y);
            float p2 = prob(m4.z, a4.z, n4.z);
            float p3 = prob(m4.w, a4.w, n4.w);
            if (j < 2) {
                pv0[(j & 1) * 4 + 0] = (_Float16)p0;
                pv0[(j & 1) * 4 + 1] = (_Float16)p1;
                pv0[(j & 1) * 4 + 2] = (_Float16)p2;
                pv0[(j & 1) * 4 + 3] = (_Float16)p3;
            } else {
                pv1[(j & 1) * 4 + 0] = (_Float16)p0;
                pv1[(j & 1) * 4 + 1] = (_Float16)p1;
                pv1[(j & 1) * 4 + 2] = (_Float16)p2;
                pv1[(j & 1) * 4 + 3] = (_Float16)p3;
            }
        }
        *(v8h*)(dst)     = pv0;
        *(v8h*)(dst + 8) = pv1;
    };

    v8f acc = {};

    produce(0);
    __syncthreads();

    for (int c = 0; c < 16; ++c) {
        const int buf = c & 1;
        const int kbase = c << 9;
        #pragma unroll
        for (int kb = 0; kb < 16; ++kb) {
            // A (16x32 f16): lane half g -> K=16g..16g+15 contiguous
            const _Float16* ap = &P[buf][r][kb * 32 + g * 16];
            v8h a0 = *(const v8h*)ap;
            v8h a1 = *(const v8h*)(ap + 8);
            v16h A;
            #pragma unroll
            for (int i = 0; i < 8; ++i) { A[i] = a0[i]; A[i + 8] = a1[i]; }
            // B (32x16 f16): lane half g -> K=16g..16g+15 contiguous in hT
            const _Float16* bp = hTc + kbase + kb * 32 + g * 16;
            v8h b0 = *(const v8h*)bp;
            v8h b1 = *(const v8h*)(bp + 8);
            v16h B;
            #pragma unroll
            for (int i = 0; i < 8; ++i) { B[i] = b0[i]; B[i + 8] = b1[i]; }
            acc = __builtin_amdgcn_wmma_f32_16x16x32_f16(
                false, A, false, B, (short)0, acc, false, false);
        }
        if (c < 15) produce(c + 1);
        __syncthreads();
    }

    // epilogue: elu, store f32
    #pragma unroll
    for (int v = 0; v < 8; ++v) {
        const int rr = row_base + v + (g << 3);
        float xv = acc[v];
        float e = xv > 0.0f ? xv : (__expf(xv) - 1.0f);
        out[(size_t)rr * OUT_F + wv * 16 + r] = e;
    }
}

// ---------------------------------------------------------------------------
extern "C" void kernel_launch(void* const* d_in, const int* in_sizes, int n_in,
                              void* d_out, int out_size, void* d_ws, size_t ws_size,
                              hipStream_t stream) {
    const float* x        = (const float*)d_in[0];
    const float* adj      = (const float*)d_in[1];
    const float* M        = (const float*)d_in[2];
    const float* W        = (const float*)d_in[3];
    const float* a_self   = (const float*)d_in[4];
    const float* a_neighs = (const float*)d_in[5];
    float* out = (float*)d_out;

    char* ws = (char*)d_ws;
    float*    h    = (float*)ws;                              // 8 MB
    _Float16* hT   = (_Float16*)(ws + (size_t)8  * 1024 * 1024); // 4 MB
    float*    as_v = (float*)   (ws + (size_t)12 * 1024 * 1024); // 32 KB
    float*    an_v = as_v + N_NODES;
    float*    rmax = an_v + N_NODES;
    float*    rinv = rmax + N_NODES;

    k_gemm_h<<<N_NODES / 16, 512, 0, stream>>>(x, W, h, hT);
    k_attn  <<<N_NODES / 8,  256, 0, stream>>>(h, a_self, a_neighs, as_v, an_v);
    k_stats <<<N_NODES,      256, 0, stream>>>(M, adj, as_v, an_v, rmax, rinv);
    k_fused <<<N_NODES / 16, 512, 0, stream>>>(M, adj, hT, as_v, an_v,
                                               rmax, rinv, out);
}
